// GCN_1116691497086
// MI455X (gfx1250) — compile-verified, matrix operand order checked by reading, not run on hardware
//
#include <hip/hip_runtime.h>
#include <hip/hip_bf16.h>

// ---------------------------------------------------------------------------
// GCN forward on MI455X (gfx1250, wave32).
// GEMMs: v_wmma_f32_16x16x32_bf16, one wave32 per 16x64 output tile
//        (4 accumulators share one A fragment), K-loop fully unrolled.
// Aggregation: non-returning global_atomic_add_f32 into L2-resident buffers.
// ---------------------------------------------------------------------------

typedef __attribute__((ext_vector_type(16))) __bf16 v16bf;
typedef __attribute__((ext_vector_type(8)))  float  v8f;

#define N_NODES 100000
#define N_EDGES 800000

// ---------------- zero fill ----------------
__global__ void k_zero(float* __restrict__ p, long long n) {
    long long i = (long long)blockIdx.x * blockDim.x + threadIdx.x;
    if (i < n) p[i] = 0.0f;
}

// ---------------- degree (atomic count of dst) ----------------
__global__ void k_degree(const int* __restrict__ dst, float* __restrict__ deg, int E) {
    int e = blockIdx.x * blockDim.x + threadIdx.x;
    if (e < E) atomicAdd(&deg[dst[e]], 1.0f);
}

// ---------------- dinv = rsqrt(deg + 1) ----------------
__global__ void k_dinv(const float* __restrict__ deg, float* __restrict__ dinv, int n) {
    int i = blockIdx.x * blockDim.x + threadIdx.x;
    if (i < n) dinv[i] = rsqrtf(deg[i] + 1.0f);
}

// ---------------- Y[M,N] = X[M,K] @ W[K,N]  (bf16 WMMA, f32 accum) ----------
// One wave32 computes a 16x64 tile: 4 x v_wmma_f32_16x16x32_bf16 per K-step,
// sharing a single A fragment. K % 32 == 0, N % 64 == 0, M % 16 == 0.
template <int K, int N>
__global__ void k_gemm_bf16(const float* __restrict__ X, const float* __restrict__ W,
                            float* __restrict__ Y, int M) {
    const int lane   = threadIdx.x & 31;
    const int waveId = blockIdx.x * (blockDim.x >> 5) + (threadIdx.x >> 5);
    constexpr int tilesN4 = N / 64;                 // 16x64 tiles per row band
    const int tileM  = waveId / tilesN4;
    const int tileN4 = waveId - tileM * tilesN4;
    if (tileM >= (M >> 4)) return;                  // wave-uniform: EXEC all-1s inside

    const int half    = lane >> 4;                  // 0: lanes 0-15, 1: lanes 16-31
    const int l16     = lane & 15;
    const int row     = (tileM << 4) + l16;         // A row held by this lane
    const int colBase = tileN4 * 64 + l16;          // first of 4 B/D columns

    const float* __restrict__ xrow = X + (long long)row * K;

    v8f acc0 = {}, acc1 = {}, acc2 = {}, acc3 = {};
#pragma unroll
    for (int k0 = 0; k0 < K; k0 += 32) {
        // A fragment (16-bit layout, ISA 7.12.2):
        //   e in [0,8):  K = k0 + half*8 + e
        //   e in [8,16): K = k0 + 16 + half*8 + (e-8)
        const int kbase = k0 + (half << 3);
        v16bf a;
#pragma unroll
        for (int e = 0; e < 8; ++e) {
            a[e]     = (__bf16)xrow[kbase + e];          // hw cvt f32->bf16
            a[e + 8] = (__bf16)xrow[kbase + 16 + e];
        }
        // 4 B fragments (cols colBase + {0,16,32,48}), same K mapping
        v16bf b0, b1, b2, b3;
#pragma unroll
        for (int e = 0; e < 16; ++e) {
            const int kk = k0 + ((e >> 3) << 4) + (half << 3) + (e & 7);
            const float* __restrict__ wrow = W + (long long)kk * N + colBase;
            b0[e] = (__bf16)wrow[0];
            b1[e] = (__bf16)wrow[16];
            b2[e] = (__bf16)wrow[32];
            b3[e] = (__bf16)wrow[48];
        }
        acc0 = __builtin_amdgcn_wmma_f32_16x16x32_bf16(false, a, false, b0, (short)0, acc0, false, false);
        acc1 = __builtin_amdgcn_wmma_f32_16x16x32_bf16(false, a, false, b1, (short)0, acc1, false, false);
        acc2 = __builtin_amdgcn_wmma_f32_16x16x32_bf16(false, a, false, b2, (short)0, acc2, false, false);
        acc3 = __builtin_amdgcn_wmma_f32_16x16x32_bf16(false, a, false, b3, (short)0, acc3, false, false);
    }

    // D layout: VGPR r -> M = r + 8*half, N = lane&15 (+ tile col offset)
#pragma unroll
    for (int r = 0; r < 8; ++r) {
        const int orow = (tileM << 4) + r + (half << 3);
        float* __restrict__ yp = Y + (long long)orow * N + colBase;
        yp[0]  = acc0[r];
        yp[16] = acc1[r];
        yp[32] = acc2[r];
        yp[48] = acc3[r];
    }
}

// ---------------- edge aggregation: agg[dst] += coef * xt[src] -------------
// One thread handles one edge x 4 consecutive features.
__global__ void k_aggregate(const float* __restrict__ xt,
                            const int* __restrict__ src, const int* __restrict__ dst,
                            const float* __restrict__ dinv, float* __restrict__ agg,
                            int E, int F) {
    const int chunks = F >> 2;
    long long idx = (long long)blockIdx.x * blockDim.x + threadIdx.x;
    long long total = (long long)E * chunks;
    if (idx >= total) return;
    const int e = (int)(idx / chunks);
    const int c = (int)(idx - (long long)e * chunks) << 2;
    const int s = src[e], d = dst[e];
    const float coef = dinv[s] * dinv[d];
    const float4 v = *(const float4*)(xt + (long long)s * F + c);
    float* ap = agg + (long long)d * F + c;
    atomicAdd(ap + 0, v.x * coef);
    atomicAdd(ap + 1, v.y * coef);
    atomicAdd(ap + 2, v.z * coef);
    atomicAdd(ap + 3, v.w * coef);
}

// -------- h = [relu](agg + dinv^2 * xt + bias), written in place into agg --
__global__ void k_self_bias_relu(const float* __restrict__ xt, float* __restrict__ agg,
                                 const float* __restrict__ dinv, const float* __restrict__ bias,
                                 int n, int F, int do_relu) {
    long long idx = (long long)blockIdx.x * blockDim.x + threadIdx.x;
    if (idx >= (long long)n * F) return;
    const int i = (int)(idx / F);
    const int f = (int)(idx - (long long)i * F);
    const float di = dinv[i];
    float v = agg[idx] + di * di * xt[idx] + bias[f];
    agg[idx] = do_relu ? fmaxf(v, 0.0f) : v;
}

// ---------------- head: out[i] = h[i,:] . Wl + bl   (F = 64) ---------------
// One wave32 per node; 2 features per lane + shuffle reduction.
__global__ void k_final(const float* __restrict__ h, const float* __restrict__ Wl,
                        const float* __restrict__ bl, float* __restrict__ out, int n) {
    const int lane = threadIdx.x & 31;
    const int node = blockIdx.x * (blockDim.x >> 5) + (threadIdx.x >> 5);
    if (node >= n) return;
    const float* hp = h + (long long)node * 64;
    float s = hp[lane] * Wl[lane] + hp[lane + 32] * Wl[lane + 32];
    for (int off = 16; off > 0; off >>= 1)
        s += __shfl_down(s, off, 32);
    if (lane == 0) out[node] = s + bl[0];
}

// ---------------------------------------------------------------------------
extern "C" void kernel_launch(void* const* d_in, const int* in_sizes, int n_in,
                              void* d_out, int out_size, void* d_ws, size_t ws_size,
                              hipStream_t stream) {
    const float* x  = (const float*)d_in[0];
    const int*   ei = (const int*)d_in[1];           // [2, E]: row0 = src, row1 = dst
    const float* W1 = (const float*)d_in[2];
    const float* b1 = (const float*)d_in[3];
    const float* W2 = (const float*)d_in[4];
    const float* b2 = (const float*)d_in[5];
    const float* W3 = (const float*)d_in[6];
    const float* b3 = (const float*)d_in[7];
    const float* Wl = (const float*)d_in[8];
    const float* bl = (const float*)d_in[9];
    float* out = (float*)d_out;

    const int N = N_NODES, E = N_EDGES;
    const int* src = ei;
    const int* dst = ei + E;

    // Workspace layout (floats), 256-float aligned offsets:
    float* ws    = (float*)d_ws;
    float* dinv  = ws;                               // N floats (padded to 100096)
    float* bufXT = ws + 100096;                      // N*256 floats (xt, also deg scratch)
    float* bufA  = bufXT + (long long)N * 256;       // N*256 floats
    float* bufB  = bufA  + (long long)N * 256;       // N*128 floats

    const int BLK = 256;
    auto blocks = [](long long n, int b) { return (unsigned)((n + b - 1) / b); };

    // --- degree -> dinv (deg scratch lives in bufXT; consumed before GEMM1) ---
    k_zero<<<blocks(N, BLK), BLK, 0, stream>>>(bufXT, N);
    k_degree<<<blocks(E, BLK), BLK, 0, stream>>>(dst, bufXT, E);
    k_dinv<<<blocks(N, BLK), BLK, 0, stream>>>(bufXT, dinv, N);

    auto gemmWaves = [&](int Nn) {
        return (long long)(N / 16) * (Nn / 64) * 32;   // threads: one wave per 16x64 tile
    };
    auto post = [&](const float* W, const float* b, float* agg, int F_out, int do_relu) {
        k_zero<<<blocks((long long)N * F_out, BLK), BLK, 0, stream>>>(agg, (long long)N * F_out);
        k_aggregate<<<blocks((long long)E * (F_out >> 2), BLK), BLK, 0, stream>>>(
            bufXT, src, dst, dinv, agg, E, F_out);
        k_self_bias_relu<<<blocks((long long)N * F_out, BLK), BLK, 0, stream>>>(
            bufXT, agg, dinv, b, N, F_out, do_relu);
        (void)W;
    };

    // Layer 1: x[N,128] @ W1[128,256] -> h1 in bufA
    k_gemm_bf16<128, 256><<<blocks(gemmWaves(256), BLK), BLK, 0, stream>>>(x, W1, bufXT, N);
    post(W1, b1, bufA, 256, 1);

    // Layer 2: h1[N,256] @ W2[256,128] -> h2 in bufB
    k_gemm_bf16<256, 128><<<blocks(gemmWaves(128), BLK), BLK, 0, stream>>>(bufA, W2, bufXT, N);
    post(W2, b2, bufB, 128, 1);

    // Layer 3: h2[N,128] @ W3[128,64] -> h3 in bufA
    k_gemm_bf16<128, 64><<<blocks(gemmWaves(64), BLK), BLK, 0, stream>>>(bufB, W3, bufXT, N);
    post(W3, b3, bufA, 64, 1);

    // head: 8 waves (nodes) per 256-thread block
    k_final<<<blocks((long long)N * 32, BLK), BLK, 0, stream>>>(bufA, Wl, bl, out, N);
}
